// PointNet2SSG_63256278335592
// MI455X (gfx1250) — compile-verified, matrix operand order checked by reading, not run on hardware
//
#include <hip/hip_runtime.h>
#include <hip/hip_bf16.h>

// PointNet++ SSG forward for MI455X (gfx1250, wave32).
//
// Conv/einsum layers -> f16-in / f32-accumulate WMMA GEMMs
// (v_wmma_f32_16x16x32_f16). Activations point-major [P, Kpad] so A/B
// fragments are contiguous 16B loads matching CDNA5 16-bit WMMA layouts.
// Per-block A tile staged into LDS with global_load_async_to_lds_b128
// (ASYNCcnt path), BN sum/sumsq fused into the GEMM epilogue via
// global_atomic_add_f32, Y stored as f16 to halve the dominant HBM traffic.

typedef __attribute__((ext_vector_type(16))) _Float16 v16h;
typedef __attribute__((ext_vector_type(8)))  _Float16 v8h;
typedef __attribute__((ext_vector_type(8)))  float    v8f;

#define TB 256

// ---------------------------------------------------------------- FPS
__global__ void __launch_bounds__(TB)
fps_kernel(const float* __restrict__ coords, int* __restrict__ out, int N, int M)
{
    __shared__ float mind[8192];
    __shared__ float rv[TB];
    __shared__ int   ri[TB];
    __shared__ int   s_last;
    const int b = blockIdx.x;
    const int t = threadIdx.x;
    const float* cb = coords + (size_t)b * 3 * N;
    for (int i = t; i < N; i += TB) mind[i] = 1e10f;
    if (t == 0) { s_last = 0; out[(size_t)b * M] = 0; }
    __syncthreads();
    for (int it = 1; it < M; ++it) {
        const int last = s_last;
        const float lx = cb[0 * N + last], ly = cb[1 * N + last], lz = cb[2 * N + last];
        float best = -1.0f; int bi = 0;
        for (int i = t; i < N; i += TB) {
            float dx = cb[0 * N + i] - lx;
            float dy = cb[1 * N + i] - ly;
            float dz = cb[2 * N + i] - lz;
            float d = dx * dx + dy * dy + dz * dz;
            float mn = fminf(mind[i], d);
            mind[i] = mn;
            if (mn > best) { best = mn; bi = i; }
        }
        rv[t] = best; ri[t] = bi;
        __syncthreads();
        for (int s = TB / 2; s > 0; s >>= 1) {
            if (t < s && rv[t + s] > rv[t]) { rv[t] = rv[t + s]; ri[t] = ri[t + s]; }
            __syncthreads();
        }
        if (t == 0) { s_last = ri[0]; out[(size_t)b * M + it] = ri[0]; }
        __syncthreads();
    }
}

// ------------------------------------------------------- gather centers
__global__ void gather_pts(const float* __restrict__ coords, int N,
                           const int* __restrict__ idx, int M,
                           float* __restrict__ out, int B)
{
    int u = blockIdx.x * blockDim.x + threadIdx.x;
    if (u >= B * M) return;
    int b = u / M, m = u % M;
    int i = idx[(size_t)b * M + m];
    for (int d = 0; d < 3; ++d)
        out[((size_t)b * 3 + d) * M + m] = coords[((size_t)b * 3 + d) * N + i];
}

// --------------------- ball query + group + pack (one wave32 per center)
// Preserves "first S in-ball indices" order via ballot + prefix popcount.
__global__ void __launch_bounds__(TB)
ball_group(const float* __restrict__ coords, int N,
           const float* __restrict__ cents, int M,
           const float* __restrict__ feats, int C,
           int S, float r2, int Cpad,
           _Float16* __restrict__ Xt, int B)
{
    const int wid  = (blockIdx.x * blockDim.x + threadIdx.x) >> 5;
    const int lane = threadIdx.x & 31;
    if (wid >= B * M) return;               // wave-uniform exit
    const int b = wid / M, m = wid % M;
    const float* cb = coords + (size_t)b * 3 * N;
    const float* fb = feats ? feats + (size_t)b * C * N : nullptr;
    const float cx = cents[((size_t)b * 3 + 0) * M + m];
    const float cy = cents[((size_t)b * 3 + 1) * M + m];
    const float cz = cents[((size_t)b * 3 + 2) * M + m];
    _Float16* base = Xt + ((size_t)b * M + m) * S * Cpad;

    int cnt = 0, firstIdx = -1;
    for (int b0 = 0; b0 < N && cnt < S; b0 += 32) {
        const int i = b0 + lane;
        bool in = false;
        float dx = 0.0f, dy = 0.0f, dz = 0.0f;
        if (i < N) {
            dx = cb[0 * N + i] - cx;
            dy = cb[1 * N + i] - cy;
            dz = cb[2 * N + i] - cz;
            in = (dx * dx + dy * dy + dz * dz) < r2;
        }
        unsigned mask = __builtin_amdgcn_ballot_w32(in);
        int slot = cnt + __popc(mask & ((1u << lane) - 1u));
        if (in && slot < S) {
            _Float16* row = base + (size_t)slot * Cpad;
            row[0] = (_Float16)dx; row[1] = (_Float16)dy; row[2] = (_Float16)dz;
            for (int c = 0; c < C; ++c) row[3 + c] = (_Float16)fb[(size_t)c * N + i];
            for (int c = 3 + C; c < Cpad; ++c) row[c] = (_Float16)0.0f;
        }
        if (firstIdx < 0 && mask) firstIdx = b0 + (int)__builtin_ctz(mask);
        cnt += __popc(mask);
    }
    if (cnt > S) cnt = S;
    if (firstIdx < 0) firstIdx = 0;
    const float fx = cb[0 * N + firstIdx] - cx;
    const float fy = cb[1 * N + firstIdx] - cy;
    const float fz = cb[2 * N + firstIdx] - cz;
    for (int s = cnt + lane; s < S; s += 32) {   // pad with first in-ball point
        _Float16* row = base + (size_t)s * Cpad;
        row[0] = (_Float16)fx; row[1] = (_Float16)fy; row[2] = (_Float16)fz;
        for (int c = 0; c < C; ++c) row[3 + c] = (_Float16)fb[(size_t)c * N + firstIdx];
        for (int c = 3 + C; c < Cpad; ++c) row[c] = (_Float16)0.0f;
    }
}

// ------------------- weights f32 -> f16 pad (+ zero BN accumulators)
__global__ void conv_w(const float* __restrict__ W, _Float16* __restrict__ Wh,
                       int O, int Cin, int Kpad,
                       float* __restrict__ sumv, float* __restrict__ sumsq)
{
    int u = blockIdx.x * blockDim.x + threadIdx.x;
    if (u < O) { sumv[u] = 0.0f; sumsq[u] = 0.0f; }
    if (u >= O * Kpad) return;
    int o = u / Kpad, k = u % Kpad;
    Wh[u] = (k < Cin) ? (_Float16)W[(size_t)o * Cin + k] : (_Float16)0.0f;
}

// --------------------------------------------------------- WMMA GEMM
// Yh[O,P](f16) = A[O,Kpad] * Xt[P,Kpad]^T, BN sum/sumsq fused via atomics.
// A tile (16 x Kpad, shared by all 4 waves) staged into LDS with async copy.
__global__ void __launch_bounds__(128)
gemm_f16f32_wmma(const _Float16* __restrict__ A,
                 const _Float16* __restrict__ Xt,
                 _Float16* __restrict__ Yh,
                 float* __restrict__ sumv, float* __restrict__ sumsq,
                 int O, int Kpad, int P)
{
    __shared__ __align__(16) _Float16 As[16 * 512];   // Kpad <= 512 -> 16KB
    __shared__ float redS[128 * 8];
    __shared__ float redQ[128 * 8];

    const int tid    = threadIdx.x;
    const int lane   = tid & 31;
    const int wave   = tid >> 5;
    const int l16    = lane & 15;
    const int half16 = lane >> 4;
    const int kb     = half16 * 8;
    const int mBase  = blockIdx.y * 16;
    const int nBase  = blockIdx.x * 256 + wave * 64;

    // ---- async stage A tile into LDS (16*Kpad*2 bytes in b128 chunks)
    {
        const int nChunks = (Kpad * 32) / 16;            // 2*Kpad
        const char* Ab = (const char*)(A + (size_t)mBase * Kpad);
        const unsigned ldsBase = (unsigned)(uintptr_t)(&As[0]);
        for (int c = tid; c < nChunks; c += 128) {
            unsigned ldsOff = ldsBase + (unsigned)(c * 16);
            unsigned long long ga = (unsigned long long)(Ab + (size_t)c * 16);
            asm volatile("global_load_async_to_lds_b128 %0, %1, off"
                         :: "v"(ldsOff), "v"(ga) : "memory");
        }
        asm volatile("s_wait_asynccnt 0x0" ::: "memory");
    }
    __syncthreads();

    v8f acc[4];
#pragma unroll
    for (int s = 0; s < 4; ++s) acc[s] = (v8f){};

    const _Float16* ArowL = As + l16 * Kpad;
    for (int k0 = 0; k0 < Kpad; k0 += 32) {
        // A fragment from LDS (ds_load_b128)
        v8h alo = *(const v8h*)(ArowL + k0 + kb);
        v8h ahi = *(const v8h*)(ArowL + k0 + 16 + kb);
        v16h a;
#pragma unroll
        for (int i = 0; i < 8; ++i) { a[i] = alo[i]; a[8 + i] = ahi[i]; }

#pragma unroll
        for (int s = 0; s < 4; ++s) {
            const _Float16* Brow = Xt + (size_t)(nBase + s * 16 + l16) * Kpad;
            __builtin_prefetch(Brow + k0 + 64, 0, 1);    // global_prefetch_b8
            v8h blo = *(const v8h*)(Brow + k0 + kb);
            v8h bhi = *(const v8h*)(Brow + k0 + 16 + kb);
            v16h bf;
#pragma unroll
            for (int i = 0; i < 8; ++i) { bf[i] = blo[i]; bf[8 + i] = bhi[i]; }
            acc[s] = __builtin_amdgcn_wmma_f32_16x16x32_f16(
                false, a, false, bf, (short)0, acc[s], false, false);
        }
    }

    // ---- store Y (f16) + per-lane BN partials
    float ps[8], pq[8];
#pragma unroll
    for (int v = 0; v < 8; ++v) { ps[v] = 0.0f; pq[v] = 0.0f; }
#pragma unroll
    for (int s = 0; s < 4; ++s) {
        int n = nBase + s * 16 + l16;
#pragma unroll
        for (int v = 0; v < 8; ++v) {
            float val = acc[s][v];
            int m = mBase + half16 * 8 + v;
            Yh[(size_t)m * P + n] = (_Float16)val;
            ps[v] += val; pq[v] += val * val;
        }
    }
#pragma unroll
    for (int v = 0; v < 8; ++v) { redS[tid * 8 + v] = ps[v]; redQ[tid * 8 + v] = pq[v]; }
    __syncthreads();
    if (tid < 16) {
        const int vv = (tid < 8) ? tid : tid - 8;
        const int lo = (tid < 8) ? 0 : 16;
        float s = 0.0f, q = 0.0f;
        for (int w = 0; w < 4; ++w)
            for (int l = 0; l < 16; ++l) {
                int t2 = w * 32 + lo + l;
                s += redS[t2 * 8 + vv];
                q += redQ[t2 * 8 + vv];
            }
        atomicAdd(&sumv[mBase + tid], s);
        atomicAdd(&sumsq[mBase + tid], q);
    }
}

// ------------------------------------------------------ BN finalize
__global__ void bn_finalize(const float* __restrict__ sumv, const float* __restrict__ sumsq,
                            int O, float Pf,
                            float* __restrict__ mean, float* __restrict__ rstd)
{
    int o = blockIdx.x * blockDim.x + threadIdx.x;
    if (o >= O) return;
    float m = sumv[o] / Pf;
    float var = sumsq[o] / Pf - m * m;
    mean[o] = m;
    rstd[o] = rsqrtf(fmaxf(var, 0.0f) + 1e-5f);
}

// ------------------------- BN + ReLU + repack to next layer f16 input
__global__ void bn_norm_pack(const _Float16* __restrict__ Y, int O, int P,
                             const float* __restrict__ mean, const float* __restrict__ rstd,
                             const float* __restrict__ g, const float* __restrict__ bt,
                             _Float16* __restrict__ Xt, int CpadNext)
{
    int p = blockIdx.x * blockDim.x + threadIdx.x;
    if (p >= P) return;
    _Float16* row = Xt + (size_t)p * CpadNext;
    for (int o = 0; o < O; ++o) {
        float v = ((float)Y[(size_t)o * P + p] - mean[o]) * rstd[o] * g[o] + bt[o];
        row[o] = (_Float16)fmaxf(v, 0.0f);
    }
    for (int o = O; o < CpadNext; ++o) row[o] = (_Float16)0.0f;
}

// --------------------- BN + ReLU + max over group dim S -> f [B, O, M]
__global__ void bn_norm_maxS(const _Float16* __restrict__ Y, int O, int B, int M, int S,
                             const float* __restrict__ mean, const float* __restrict__ rstd,
                             const float* __restrict__ g, const float* __restrict__ bt,
                             float* __restrict__ f)
{
    int u = blockIdx.x * blockDim.x + threadIdx.x;
    if (u >= B * O * M) return;
    int b = u / (O * M);
    int o = (u / M) % O;
    int m = u % M;
    const int P = B * M * S;
    const float mu = mean[o], rs = rstd[o], gg = g[o], bb = bt[o];
    const _Float16* row = Y + (size_t)o * P + ((size_t)b * M + m) * S;
    float mx = 0.0f;                        // relu output >= 0
    for (int s = 0; s < S; ++s) {
        float v = fmaxf(((float)row[s] - mu) * rs * gg + bb, 0.0f);
        mx = fmaxf(mx, v);
    }
    f[((size_t)b * O + o) * M + m] = mx;
}

// ---------------- BN + ReLU + max over points (global pool) -> [B, O]
__global__ void bn_norm_maxN(const _Float16* __restrict__ Y, int O, int B, int Nn,
                             const float* __restrict__ mean, const float* __restrict__ rstd,
                             const float* __restrict__ g, const float* __restrict__ bt,
                             float* __restrict__ out)
{
    int u = blockIdx.x * blockDim.x + threadIdx.x;
    if (u >= B * O) return;
    int b = u / O, o = u % O;
    const int P = B * Nn;
    const float mu = mean[o], rs = rstd[o], gg = g[o], bb = bt[o];
    const _Float16* row = Y + (size_t)o * P + (size_t)b * Nn;
    float mx = 0.0f;
    for (int n = 0; n < Nn; ++n) {
        float v = fmaxf(((float)row[n] - mu) * rs * gg + bb, 0.0f);
        mx = fmaxf(mx, v);
    }
    out[(size_t)b * O + o] = mx;
}

// ------------------------------------- global stage input pack
__global__ void pack_global(const float* __restrict__ pt2, const float* __restrict__ f2,
                            int B, int Nn, int C, int Cpad, _Float16* __restrict__ Xt)
{
    int u = blockIdx.x * blockDim.x + threadIdx.x;
    if (u >= B * Nn) return;
    int b = u / Nn, n = u % Nn;
    _Float16* row = Xt + (size_t)u * Cpad;
    for (int d = 0; d < 3; ++d) row[d] = (_Float16)pt2[((size_t)b * 3 + d) * Nn + n];
    for (int c = 0; c < C; ++c) row[3 + c] = (_Float16)f2[((size_t)b * C + c) * Nn + n];
    for (int c = 3 + C; c < Cpad; ++c) row[c] = (_Float16)0.0f;
}

// --------------------------------------------------------- FC head
__global__ void fc_fwd(const float* __restrict__ X, const float* __restrict__ W,
                       const float* __restrict__ bias, float* __restrict__ Yo,
                       int B, int Cin, int O)
{
    int u = blockIdx.x * blockDim.x + threadIdx.x;
    if (u >= B * O) return;
    int b = u / O, o = u % O;
    float s = bias ? bias[o] : 0.0f;
    const float* x = X + (size_t)b * Cin;
    const float* w = W + (size_t)o * Cin;
    for (int c = 0; c < Cin; ++c) s += x[c] * w[c];
    Yo[u] = s;
}

__global__ void bn_batch_relu(float* __restrict__ X, int B, int O,
                              const float* __restrict__ g, const float* __restrict__ bt)
{
    int o = blockIdx.x * blockDim.x + threadIdx.x;
    if (o >= O) return;
    float s = 0.0f, sq = 0.0f;
    for (int b = 0; b < B; ++b) { float v = X[(size_t)b * O + o]; s += v; sq += v * v; }
    float m = s / B, var = sq / B - m * m;
    float rs = rsqrtf(fmaxf(var, 0.0f) + 1e-5f);
    for (int b = 0; b < B; ++b) {
        float v = (X[(size_t)b * O + o] - m) * rs * g[o] + bt[o];
        X[(size_t)b * O + o] = fmaxf(v, 0.0f);
    }
}

// ================================================================ host
static inline int nb(int n, int b = TB) { return (n + b - 1) / b; }

extern "C" void kernel_launch(void* const* d_in, const int* in_sizes, int n_in,
                              void* d_out, int out_size, void* d_ws, size_t ws_size,
                              hipStream_t stream)
{
    // Input order: pt_coordinates, then params flattened with sorted dict keys
    // (fc1, fc2, fc3, ssg1, ssg2, ssg3; within each: W, b, bt, g).
    const float* pt = (const float*)d_in[0];
    int i = 1;
    const float *fc1W = (const float*)d_in[i++], *fc1b = (const float*)d_in[i++],
                *fc1bt = (const float*)d_in[i++], *fc1g = (const float*)d_in[i++];
    const float *fc2W = (const float*)d_in[i++], *fc2b = (const float*)d_in[i++],
                *fc2bt = (const float*)d_in[i++], *fc2g = (const float*)d_in[i++];
    const float *fc3W = (const float*)d_in[i++], *fc3b = (const float*)d_in[i++];
    const float *sW[3][3], *sb[3][3], *sbt[3][3], *sg[3][3];
    for (int st = 0; st < 3; ++st)
        for (int l = 0; l < 3; ++l) {
            sW[st][l]  = (const float*)d_in[i++];
            sb[st][l]  = (const float*)d_in[i++];
            sbt[st][l] = (const float*)d_in[i++];
            sg[st][l]  = (const float*)d_in[i++];
        }
    (void)fc1b; (void)fc2b; (void)sb;   // bias before BN cancels

    const int B = 32, N0 = 8192, M1 = 512, S1 = 32, M2 = 128, S2 = 64, N3 = 128;
    const int P1 = B * M1 * S1;   // 524288
    const int P2 = B * M2 * S2;   // 262144
    const int P3 = B * N3;        // 4096

    char* ws = (char*)d_ws;
    size_t off = 0;
    auto alloc = [&](size_t bytes) -> void* {
        void* p = ws + off;
        off = (off + bytes + 255) & ~(size_t)255;
        return p;
    };
    int*      c1    = (int*)alloc((size_t)B * M1 * 4);
    int*      c2    = (int*)alloc((size_t)B * M2 * 4);
    float*    pt1   = (float*)alloc((size_t)B * 3 * M1 * 4);
    float*    pt2   = (float*)alloc((size_t)B * 3 * N3 * 4);
    _Float16* Wh    = (_Float16*)alloc((size_t)1024 * 512 * 2);
    float*    mean  = (float*)alloc(1024 * 4);
    float*    rstd  = (float*)alloc(1024 * 4);
    float*    sumv  = (float*)alloc(1024 * 4);
    float*    sumsq = (float*)alloc(1024 * 4);
    float*    f1    = (float*)alloc((size_t)B * 128 * M1 * 4);
    float*    f2    = (float*)alloc((size_t)B * 256 * N3 * 4);
    float*    g3    = (float*)alloc((size_t)B * 1024 * 4);
    float*    y1    = (float*)alloc((size_t)B * 512 * 4);
    float*    y2    = (float*)alloc((size_t)B * 256 * 4);
    _Float16* X     = (_Float16*)alloc((size_t)P2 * 160 * 2);       // max X
    _Float16* Y     = (_Float16*)alloc((size_t)256 * (size_t)P2 * 2); // max Y (f16)
    (void)ws_size;

    auto layer = [&](const float* W, const float* gam, const float* bet,
                     int O, int Cin, int Kpad, int P, int CpadNext) {
        conv_w<<<nb(O * Kpad), TB, 0, stream>>>(W, Wh, O, Cin, Kpad, sumv, sumsq);
        gemm_f16f32_wmma<<<dim3(P / 256, O / 16), 128, 0, stream>>>(Wh, X, Y,
                                                                    sumv, sumsq, O, Kpad, P);
        bn_finalize<<<nb(O), TB, 0, stream>>>(sumv, sumsq, O, (float)P, mean, rstd);
        if (CpadNext > 0)
            bn_norm_pack<<<nb(P), TB, 0, stream>>>(Y, O, P, mean, rstd, gam, bet, X, CpadNext);
    };

    // ---------------- stage 1
    fps_kernel<<<B, TB, 0, stream>>>(pt, c1, N0, M1);
    gather_pts<<<nb(B * M1), TB, 0, stream>>>(pt, N0, c1, M1, pt1, B);
    ball_group<<<nb(B * M1 * 32), TB, 0, stream>>>(pt, N0, pt1, M1, nullptr, 0,
                                                   S1, 0.2f * 0.2f, 32, X, B);
    layer(sW[0][0], sg[0][0], sbt[0][0],  64,  3,  32, P1, 64);
    layer(sW[0][1], sg[0][1], sbt[0][1],  64, 64,  64, P1, 64);
    layer(sW[0][2], sg[0][2], sbt[0][2], 128, 64,  64, P1, 0);
    bn_norm_maxS<<<nb(B * 128 * M1), TB, 0, stream>>>(Y, 128, B, M1, S1,
                                                      mean, rstd, sg[0][2], sbt[0][2], f1);

    // ---------------- stage 2
    fps_kernel<<<B, TB, 0, stream>>>(pt1, c2, M1, M2);
    gather_pts<<<nb(B * M2), TB, 0, stream>>>(pt1, M1, c2, M2, pt2, B);
    ball_group<<<nb(B * M2 * 32), TB, 0, stream>>>(pt1, M1, pt2, M2, f1, 128,
                                                   S2, 0.4f * 0.4f, 160, X, B);
    layer(sW[1][0], sg[1][0], sbt[1][0], 128, 131, 160, P2, 128);
    layer(sW[1][1], sg[1][1], sbt[1][1], 128, 128, 128, P2, 128);
    layer(sW[1][2], sg[1][2], sbt[1][2], 256, 128, 128, P2, 0);
    bn_norm_maxS<<<nb(B * 256 * N3), TB, 0, stream>>>(Y, 256, B, N3, S2,
                                                      mean, rstd, sg[1][2], sbt[1][2], f2);

    // ---------------- global abstraction
    pack_global<<<nb(B * N3), TB, 0, stream>>>(pt2, f2, B, N3, 256, 288, X);
    layer(sW[2][0], sg[2][0], sbt[2][0],  256, 259, 288, P3, 256);
    layer(sW[2][1], sg[2][1], sbt[2][1],  512, 256, 256, P3, 512);
    layer(sW[2][2], sg[2][2], sbt[2][2], 1024, 512, 512, P3, 0);
    bn_norm_maxN<<<nb(B * 1024), TB, 0, stream>>>(Y, 1024, B, N3,
                                                  mean, rstd, sg[2][2], sbt[2][2], g3);

    // ---------------- FC head
    fc_fwd<<<nb(B * 512), TB, 0, stream>>>(g3, fc1W, nullptr, y1, B, 1024, 512);
    bn_batch_relu<<<nb(512), TB, 0, stream>>>(y1, B, 512, fc1g, fc1bt);
    fc_fwd<<<nb(B * 256), TB, 0, stream>>>(y1, fc2W, nullptr, y2, B, 512, 256);
    bn_batch_relu<<<nb(256), TB, 0, stream>>>(y2, B, 256, fc2g, fc2bt);
    fc_fwd<<<nb(B * 40), TB, 0, stream>>>(y2, fc3W, fc3b, (float*)d_out, B, 256, 40);
    (void)in_sizes; (void)n_in; (void)out_size;
}
